// SP_CAPS_80874234184210
// MI455X (gfx1250) — compile-verified, matrix operand order checked by reading, not run on hardware
//
#include <hip/hip_runtime.h>
#include <stdint.h>
#include <math.h>

// ---------------------------------------------------------------------------
// Types for CDNA5 WMMA (wave32, 16x16x32 bf16 -> f32)
// ---------------------------------------------------------------------------
typedef __attribute__((ext_vector_type(16))) __bf16 v16bf;
typedef __attribute__((ext_vector_type(8)))  __bf16 v8bf;
typedef __attribute__((ext_vector_type(8)))  float  v8f;

#define KTOP 2048
#define NMS_RAD 8
#define REF_RAD 3
#define BORDER_PX 4
#define ROWP 40  // LDS row pitch in bf16 (80B): 16B-aligned, bank-conflict pad

// ---------------------------------------------------------------------------
// Small prep kernels
// ---------------------------------------------------------------------------
__global__ void k_fold_bn(const float* __restrict__ g, const float* __restrict__ bb,
                          const float* __restrict__ mm, const float* __restrict__ vv,
                          const float* __restrict__ cb,
                          float* __restrict__ scale, float* __restrict__ bias, int C) {
  int c = blockIdx.x * blockDim.x + threadIdx.x;
  if (c >= C) return;
  float s = g[c] * rsqrtf(vv[c] + 1e-5f);
  scale[c] = s;
  float conv_b = cb ? cb[c] : 0.f;
  bias[c] = (conv_b - mm[c]) * s + bb[c];
}

__global__ void k_bias_only(const float* __restrict__ cb,
                            float* __restrict__ scale, float* __restrict__ bias, int C) {
  int c = blockIdx.x * blockDim.x + threadIdx.x;
  if (c >= C) return;
  scale[c] = 1.f;
  bias[c] = cb ? cb[c] : 0.f;
}

// fp32 [O][Ktot] -> bf16 [O][Kpad] (zero padded K)
__global__ void k_pack_w(const float* __restrict__ w, __bf16* __restrict__ out,
                         int O, int Ktot, int Kpad) {
  int i = blockIdx.x * blockDim.x + threadIdx.x;
  if (i >= O * Kpad) return;
  int o = i / Kpad;
  int k = i - o * Kpad;
  out[i] = (__bf16)((k < Ktot) ? w[(size_t)o * Ktot + k] : 0.f);
}

// ---------------------------------------------------------------------------
// Implicit-GEMM convolution, stride 1, fused scale/bias (+residual) (+ReLU).
//   A (M x K) = im2col(activations), bf16 in LDS
//   B (K x N) = weights [Cout][Kpad], streamed global->LDS with async b128
//   C (M x N) fp32 accum in WMMA fragments.
// Block tile 256(M) x 64(N); 8 wave32, each owns a 32x64 strip = 2x4 WMMA
// fragments => 8 v_wmma per K-step per wave.
// SPEC: 1 = pointwise (1x1), 9 = 3x3 (constant-divisor decode), 0 = generic.
// ---------------------------------------------------------------------------
template <int SPEC>
__global__ void __launch_bounds__(256)
conv_wmma(const float* __restrict__ src, const __bf16* __restrict__ wp,
          const float* __restrict__ scale, const float* __restrict__ bias,
          const float* __restrict__ res, float* __restrict__ out,
          int B, int Cin, int H, int W, int Cout,
          int KH, int KW, int pad, int Kpad, int relu) {
  __shared__ __align__(16) __bf16 Asm[2][256][ROWP];
  __shared__ __align__(16) __bf16 Bsm[2][64][ROWP];

  const int t    = threadIdx.x;
  const int lane = t & 31;
  const int wv   = t >> 5;          // 0..7
  const int wm   = wv * 32;         // wave M offset in block tile
  const int rsel = lane & 15;       // row/col within 16
  const int hsel = (lane >> 4) * 8; // K-half select (A/B frags), M-half (C)

  const int HW   = H * W;
  const int KHWc = (SPEC == 1) ? 1 : (SPEC == 9 ? 9 : KH * KW);
  const int KWc  = (SPEC == 9) ? 3 : KW;
  const int Ktot = Cin * KHWc;
  const int M    = B * HW;
  const int m0   = blockIdx.x * 256;
  const int n0   = blockIdx.y * 64;

  // A-fill mapping: thread t fills row t (adjacent threads = adjacent pixels
  // so each fixed-k gather is coalesced across the wave).
  const int am = m0 + t;
  const bool am_ok = am < M;
  int ab = 0, ay = 0, ax = 0;
  if (am_ok) { ab = am / HW; int p = am - ab * HW; ay = p / W; ax = p - ay * W; }
  const float* pwbase = src + ((size_t)ab * Cin) * HW + (size_t)ay * W + ax;

  auto fillA = [&](int bi, int k0) {
    if (SPEC == 1) {  // 1x1 conv: K == Cin, no spatial decode, no bounds checks
#pragma unroll 8
      for (int e = 0; e < 32; ++e) {
        int kg = k0 + e;
        float v = (am_ok && kg < Ktot) ? pwbase[(size_t)kg * HW] : 0.f;
        Asm[bi][t][e] = (__bf16)v;
      }
    } else {
#pragma unroll 4
      for (int e = 0; e < 32; ++e) {
        int kg = k0 + e;
        float v = 0.f;
        if (am_ok && kg < Ktot) {
          int cc = kg / KHWc;          // constant divisor when SPEC==9
          int r2 = kg - cc * KHWc;
          int kh = r2 / KWc;
          int kw = r2 - kh * KWc;
          int iy = ay + kh - pad;
          int ix = ax + kw - pad;
          if (iy >= 0 && iy < H && ix >= 0 && ix < W)
            v = src[(size_t)(ab * Cin + cc) * HW + (size_t)iy * W + ix];
        }
        Asm[bi][t][e] = (__bf16)v;
      }
    }
  };

  // B-fill: 64 rows x 4 sixteen-byte chunks = 256 threads exactly.
  const int bnr  = t >> 2;
  const int bsub = t & 3;
  const int gnb  = n0 + bnr;

  auto fillB = [&](int bi, int k0) {
    if (gnb < Cout) {
      unsigned ldsoff = (unsigned)(uintptr_t)&Bsm[bi][bnr][bsub * 8];
      unsigned long long gp =
          (unsigned long long)(uintptr_t)(wp + (size_t)gnb * Kpad + k0 + bsub * 8);
      // CDNA5 async copy global -> LDS (tracked by ASYNCcnt)
      asm volatile("global_load_async_to_lds_b128 %0, %1, off"
                   :: "v"(ldsoff), "v"(gp) : "memory");
      if (k0 + 64 < Kpad)  // lowers to global_prefetch_b8
        __builtin_prefetch((const void*)(uintptr_t)(gp + 128), 0, 1);
    } else {
      v8bf z;
#pragma unroll
      for (int i = 0; i < 8; ++i) z[i] = (__bf16)0.f;
      *(v8bf*)&Bsm[bi][bnr][bsub * 8] = z;
    }
  };

  v8f acc[2][4];
#pragma unroll
  for (int i = 0; i < 2; ++i)
#pragma unroll
    for (int j = 0; j < 4; ++j)
#pragma unroll
      for (int e = 0; e < 8; ++e) acc[i][j][e] = 0.f;

  int buf = 0;
  fillA(0, 0);
  fillB(0, 0);
  asm volatile("s_wait_asynccnt 0" ::: "memory");
  __syncthreads();

  for (int k0 = 0; k0 < Kpad; k0 += 32) {
    int nb = buf ^ 1;
    if (k0 + 32 < Kpad) { fillA(nb, k0 + 32); fillB(nb, k0 + 32); }

    // A fragment: lanes 0-15 hold rows M=0..15, K 0..7 & 16..23;
    // lanes 16-31 same rows, K 8..15 & 24..31 (ISA 16-bit A layout).
    v16bf af[2], bfg[4];
#pragma unroll
    for (int i = 0; i < 2; ++i) {
      const __bf16* pa = &Asm[buf][wm + i * 16 + rsel][0];
      v8bf lo = *(const v8bf*)(pa + hsel);
      v8bf hi = *(const v8bf*)(pa + 16 + hsel);
      af[i] = __builtin_shufflevector(lo, hi, 0, 1, 2, 3, 4, 5, 6, 7,
                                      8, 9, 10, 11, 12, 13, 14, 15);
    }
#pragma unroll
    for (int j = 0; j < 4; ++j) {
      const __bf16* pb = &Bsm[buf][j * 16 + rsel][0];
      v8bf lo = *(const v8bf*)(pb + hsel);
      v8bf hi = *(const v8bf*)(pb + 16 + hsel);
      bfg[j] = __builtin_shufflevector(lo, hi, 0, 1, 2, 3, 4, 5, 6, 7,
                                       8, 9, 10, 11, 12, 13, 14, 15);
    }
#pragma unroll
    for (int i = 0; i < 2; ++i)
#pragma unroll
      for (int j = 0; j < 4; ++j)
        acc[i][j] = __builtin_amdgcn_wmma_f32_16x16x32_bf16(
            false, af[i], false, bfg[j], (short)0, acc[i][j], false, false);

    asm volatile("s_wait_asynccnt 0" ::: "memory");
    __syncthreads();
    buf = nb;
  }

  // Epilogue: fused scale/bias (+res) (+relu). C layout: N = lane&15,
  // M = e + 8*(lane>>4). Fragment elements are M-contiguous and (HW%8==0,
  // base 8-aligned) never cross a batch boundary -> 2x b128 stores.
#pragma unroll
  for (int i = 0; i < 2; ++i) {
    int mbase = m0 + wm + i * 16 + hsel;
    if (mbase >= M) continue;
    int mb = mbase / HW;
    int p  = mbase - mb * HW;
    size_t obase = (size_t)mb * Cout * HW + p;  // + n*HW per channel
    bool full = (mbase + 7 < M);
#pragma unroll
    for (int j = 0; j < 4; ++j) {
      int n = n0 + j * 16 + rsel;
      if (n >= Cout) continue;
      float sc = scale[n], bs = bias[n];
      size_t o = obase + (size_t)n * HW;
      if (full) {
        float r[8];
#pragma unroll
        for (int e = 0; e < 8; ++e) r[e] = acc[i][j][e] * sc + bs;
        if (res) {
          float4 ra = ((const float4*)(res + o))[0];
          float4 rb = ((const float4*)(res + o))[1];
          r[0] += ra.x; r[1] += ra.y; r[2] += ra.z; r[3] += ra.w;
          r[4] += rb.x; r[5] += rb.y; r[6] += rb.z; r[7] += rb.w;
        }
        if (relu) {
#pragma unroll
          for (int e = 0; e < 8; ++e) r[e] = fmaxf(r[e], 0.f);
        }
        ((float4*)(out + o))[0] = make_float4(r[0], r[1], r[2], r[3]);
        ((float4*)(out + o))[1] = make_float4(r[4], r[5], r[6], r[7]);
      } else {
        for (int e = 0; e < 8; ++e) {
          if (mbase + e >= M) break;
          float v = acc[i][j][e] * sc + bs;
          if (res) v += res[o + e];
          if (relu) v = fmaxf(v, 0.f);
          out[o + e] = v;
        }
      }
    }
  }
}

// ---------------------------------------------------------------------------
// Pointwise / filtering kernels
// ---------------------------------------------------------------------------
__global__ void k_maxpool2(const float* __restrict__ in, float* __restrict__ out,
                           int BC, int Ho, int Wo) {
  int i = blockIdx.x * blockDim.x + threadIdx.x;
  if (i >= BC * Ho * Wo) return;
  int x = i % Wo;
  int tp = i / Wo;
  int y = tp % Ho;
  int bc = tp / Ho;
  int Wi = Wo * 2;
  const float* p = in + ((size_t)bc * (Ho * 2) + y * 2) * Wi + x * 2;
  out[i] = fmaxf(fmaxf(p[0], p[1]), fmaxf(p[Wi], p[Wi + 1]));
}

// softmax over 65 channels, drop dust bin, 8x8 pixel shuffle
__global__ void k_softmax_shuffle(const float* __restrict__ logits,
                                  float* __restrict__ dense,
                                  int B, int hc, int wc) {
  int cell = blockIdx.x * blockDim.x + threadIdx.x;
  int tot = B * hc * wc;
  if (cell >= tot) return;
  int b = cell / (hc * wc);
  int r = cell - b * hc * wc;
  int yc = r / wc;
  int xc = r - yc * wc;
  const size_t cs = (size_t)hc * wc;
  const float* p = logits + (size_t)b * 65 * cs + r;
  float mx = -1e30f;
  for (int c = 0; c < 65; ++c) mx = fmaxf(mx, p[c * cs]);
  float e[64], s = 0.f;
  for (int c = 0; c < 65; ++c) {
    float t = expf(p[c * cs] - mx);
    if (c < 64) e[c] = t;
    s += t;
  }
  float inv = 1.f / s;
  int Hd = hc * 8, Wd = wc * 8;
  for (int dy = 0; dy < 8; ++dy)
    for (int dx = 0; dx < 8; ++dx)
      dense[(size_t)b * Hd * Wd + (size_t)(yc * 8 + dy) * Wd + (xc * 8 + dx)] =
          e[dy * 8 + dx] * inv;
}

__global__ void k_l2norm(const float* __restrict__ in, float* __restrict__ out,
                         int B, int C, int HW) {
  int i = blockIdx.x * blockDim.x + threadIdx.x;
  if (i >= B * HW) return;
  int b = i / HW;
  int p = i - b * HW;
  const float* f = in + (size_t)b * C * HW + p;
  float s = 0.f;
  for (int c = 0; c < C; ++c) { float v = f[(size_t)c * HW]; s += v * v; }
  float inv = 1.f / fmaxf(sqrtf(s), 1e-12f);
  float* o = out + (size_t)b * C * HW + p;
  for (int c = 0; c < C; ++c) o[(size_t)c * HW] = f[(size_t)c * HW] * inv;
}

__global__ void k_maxfilt_h(const float* __restrict__ in, float* __restrict__ out,
                            int B, int H, int W, int r) {
  int i = blockIdx.x * blockDim.x + threadIdx.x;
  if (i >= B * H * W) return;
  int x = i % W;
  int base = i - x;
  float m = -1e30f;
  for (int d = -r; d <= r; ++d) {
    int xx = x + d;
    if (xx >= 0 && xx < W) m = fmaxf(m, in[base + xx]);
  }
  out[i] = m;
}

__global__ void k_maxfilt_v(const float* __restrict__ in, float* __restrict__ out,
                            int B, int H, int W, int r) {
  int i = blockIdx.x * blockDim.x + threadIdx.x;
  if (i >= B * H * W) return;
  int x = i % W;
  int tp = i / W;
  int y = tp % H;
  int b = tp / H;
  const float* col = in + (size_t)b * H * W + x;
  float m = -1e30f;
  for (int d = -r; d <= r; ++d) {
    int yy = y + d;
    if (yy >= 0 && yy < H) m = fmaxf(m, col[(size_t)yy * W]);
  }
  out[i] = m;
}

__global__ void k_box_h(const float* __restrict__ in, float* __restrict__ out,
                        int B, int H, int W, int r) {
  int i = blockIdx.x * blockDim.x + threadIdx.x;
  if (i >= B * H * W) return;
  int x = i % W;
  int base = i - x;
  float s = 0.f;
  for (int d = -r; d <= r; ++d) {
    int xx = x + d;
    if (xx >= 0 && xx < W) s += in[base + xx];
  }
  out[i] = s;
}

__global__ void k_box_v(const float* __restrict__ in, float* __restrict__ out,
                        int B, int H, int W, int r) {
  int i = blockIdx.x * blockDim.x + threadIdx.x;
  if (i >= B * H * W) return;
  int x = i % W;
  int tp = i / W;
  int y = tp % H;
  int b = tp / H;
  const float* col = in + (size_t)b * H * W + x;
  float s = 0.f;
  for (int d = -r; d <= r; ++d) {
    int yy = y + d;
    if (yy >= 0 && yy < H) s += col[(size_t)yy * W];
  }
  out[i] = s;
}

__global__ void k_lin_h(const float* __restrict__ in, float* __restrict__ out,
                        int B, int H, int W, int r) {
  int i = blockIdx.x * blockDim.x + threadIdx.x;
  if (i >= B * H * W) return;
  int x = i % W;
  int base = i - x;
  float s = 0.f;
  for (int d = -r; d <= r; ++d) {
    int xx = x + d;
    if (xx >= 0 && xx < W) s += in[base + xx] * (float)d;
  }
  out[i] = s;
}

__global__ void k_lin_v(const float* __restrict__ in, float* __restrict__ out,
                        int B, int H, int W, int r) {
  int i = blockIdx.x * blockDim.x + threadIdx.x;
  if (i >= B * H * W) return;
  int x = i % W;
  int tp = i / W;
  int y = tp % H;
  int b = tp / H;
  const float* col = in + (size_t)b * H * W + x;
  float s = 0.f;
  for (int d = -r; d <= r; ++d) {
    int yy = y + d;
    if (yy >= 0 && yy < H) s += col[(size_t)yy * W] * (float)d;
  }
  out[i] = s;
}

__global__ void k_eq(float* __restrict__ out, const float* __restrict__ a,
                     const float* __restrict__ b, int n) {
  int i = blockIdx.x * blockDim.x + threadIdx.x;
  if (i >= n) return;
  out[i] = (a[i] == b[i]) ? 1.f : 0.f;
}

__global__ void k_suppress(const float* __restrict__ mpm, const float* __restrict__ s,
                           float* __restrict__ supp, float* __restrict__ ss, int n) {
  int i = blockIdx.x * blockDim.x + threadIdx.x;
  if (i >= n) return;
  float sp = (mpm[i] > 0.f) ? 1.f : 0.f;
  supp[i] = sp;
  ss[i] = (sp != 0.f) ? 0.f : s[i];
}

__global__ void k_mask_update(float* __restrict__ mask, const float* __restrict__ ss,
                              const float* __restrict__ mpss,
                              const float* __restrict__ supp, int n) {
  int i = blockIdx.x * blockDim.x + threadIdx.x;
  if (i >= n) return;
  bool nm = (ss[i] == mpss[i]) && (supp[i] == 0.f);
  mask[i] = ((mask[i] != 0.f) || nm) ? 1.f : 0.f;
}

__global__ void k_border_masked(float* __restrict__ masked, const float* __restrict__ mask,
                                const float* __restrict__ s, int B, int H, int W, int bd) {
  int i = blockIdx.x * blockDim.x + threadIdx.x;
  if (i >= B * H * W) return;
  int x = i % W;
  int y = (i / W) % H;
  bool ok = (x >= bd) && (x < W - bd) && (y >= bd) && (y < H - bd);
  float v = s[i];
  masked[i] = (ok && mask[i] != 0.f && v > 0.f) ? v : -1.f;
}

__global__ void k_div_delta(float* __restrict__ dy, float* __restrict__ dx,
                            const float* __restrict__ box, int n) {
  int i = blockIdx.x * blockDim.x + threadIdx.x;
  if (i >= n) return;
  float inv = 1.f / fmaxf(box[i], 1e-6f);
  dy[i] *= inv;
  dx[i] *= inv;
}

// ---------------------------------------------------------------------------
// Top-K via ordered-float histogram
// ---------------------------------------------------------------------------
__device__ __forceinline__ unsigned fkey(float v) {
  unsigned u = __float_as_uint(v);
  return ((int)u >= 0) ? (u | 0x80000000u) : ~u;
}

__global__ void k_hist(const float* __restrict__ masked, int* __restrict__ hist,
                       int B, int HW) {
  int i = blockIdx.x * blockDim.x + threadIdx.x;
  if (i >= B * HW) return;
  int b = i / HW;
  int bin = fkey(masked[i]) >> 21;  // 2048 bins
  atomicAdd(&hist[b * 2048 + bin], 1);
}

__global__ void k_select(const int* __restrict__ hist, int* __restrict__ tb,
                         int* __restrict__ above, int B, int K) {
  int b = threadIdx.x;
  if (b >= B) return;
  const int* h = hist + b * 2048;
  int acc = 0, t = 0;
  for (int bin = 2047; bin >= 0; --bin) {
    int c = h[bin];
    if (acc + c >= K) { t = bin; break; }
    acc += c;
  }
  tb[b] = t;
  above[b] = acc;
}

__global__ void k_compact(const float* __restrict__ masked, const int* __restrict__ tb,
                          const int* __restrict__ above, int* __restrict__ cnt0,
                          int* __restrict__ cnt1, float* __restrict__ kp_scores,
                          int* __restrict__ kp_idx, int B, int HW, int K) {
  int i = blockIdx.x * blockDim.x + threadIdx.x;
  if (i >= B * HW) return;
  int b = i / HW;
  int p = i - b * HW;
  float v = masked[i];
  int bin = fkey(v) >> 21;
  int t = tb[b];
  if (bin > t) {
    int slot = atomicAdd(&cnt0[b], 1);
    if (slot < K) { kp_scores[b * K + slot] = v; kp_idx[b * K + slot] = p; }
  } else if (bin == t) {
    int slot = above[b] + atomicAdd(&cnt1[b], 1);
    if (slot < K) { kp_scores[b * K + slot] = v; kp_idx[b * K + slot] = p; }
  }
}

__global__ void k_refine(const int* __restrict__ kp_idx, const float* __restrict__ dyn,
                         const float* __restrict__ dxn, float* __restrict__ kpts_xy,
                         int B, int K, int H, int W) {
  int i = blockIdx.x * blockDim.x + threadIdx.x;
  if (i >= B * K) return;
  int b = i / K;
  int p = kp_idx[i];
  int y = p / W;
  int x = p - y * W;
  size_t o = (size_t)b * H * W + p;
  kpts_xy[(size_t)i * 2 + 0] = (float)x + dxn[o];
  kpts_xy[(size_t)i * 2 + 1] = (float)y + dyn[o];
}

__device__ __forceinline__ float tap4(const float* f, int xi, int yi, int wq, int hq) {
  if (xi < 0 || xi >= wq || yi < 0 || yi >= hq) return 0.f;
  return f[(size_t)yi * wq + xi];
}

// one 128-thread block per keypoint: bilinear sample + L2 normalize
__global__ void k_sample(const float* __restrict__ kpts, const float* __restrict__ dmap,
                         float* __restrict__ out, int B, int K, int C,
                         int hc, int wc, int H, int W) {
  int kp = blockIdx.x;
  int c = threadIdx.x;
  int b = kp / K;
  float px = kpts[(size_t)kp * 2 + 0];
  float py = kpts[(size_t)kp * 2 + 1];
  float nx = px / (float)W * 2.f - 1.f;
  float ny = py / (float)H * 2.f - 1.f;
  float ix = ((nx + 1.f) * (float)wc - 1.f) * 0.5f;
  float iy = ((ny + 1.f) * (float)hc - 1.f) * 0.5f;
  float x0f = floorf(ix), y0f = floorf(iy);
  float wx1 = ix - x0f, wy1 = iy - y0f;
  int x0 = (int)x0f, y0 = (int)y0f;
  const float* f = dmap + ((size_t)b * C + c) * hc * wc;
  float acc = tap4(f, x0, y0, wc, hc) * (1.f - wx1) * (1.f - wy1)
            + tap4(f, x0 + 1, y0, wc, hc) * wx1 * (1.f - wy1)
            + tap4(f, x0, y0 + 1, wc, hc) * (1.f - wx1) * wy1
            + tap4(f, x0 + 1, y0 + 1, wc, hc) * wx1 * wy1;
  __shared__ float sh[128];
  sh[c] = acc * acc;
  __syncthreads();
  if (c == 0) {
    float s2 = 0.f;
    for (int j = 0; j < 128; ++j) s2 += sh[j];
    sh[0] = 1.f / fmaxf(sqrtf(s2), 1e-12f);
  }
  __syncthreads();
  out[(size_t)kp * C + c] = acc * sh[0];
}

// ---------------------------------------------------------------------------
// Host orchestration
// ---------------------------------------------------------------------------
struct BNp { const float *b, *g, *m, *v; };
struct CVp { const float *b, *w; };
struct Blk { BNp bn1, bn2, bn3; CVp c1, c2, c3; CVp ds; BNp dsbn; bool has_ds; };
struct SB  { const float* scale; const float* bias; };

struct Ctx {
  hipStream_t s;
  char* ws;
  size_t off;
};

static void* wsal(Ctx& c, size_t bytes) {
  c.off = (c.off + 255) & ~(size_t)255;
  void* p = c.ws + c.off;
  c.off += bytes;
  return p;
}

static SB fold(Ctx& c, const BNp& bn, const float* cb, int C) {
  float* sc = (float*)wsal(c, (size_t)C * 4);
  float* bs = (float*)wsal(c, (size_t)C * 4);
  k_fold_bn<<<(C + 255) / 256, 256, 0, c.s>>>(bn.g, bn.b, bn.m, bn.v, cb, sc, bs, C);
  return {sc, bs};
}

static SB bias_only(Ctx& c, const float* cb, int C) {
  float* sc = (float*)wsal(c, (size_t)C * 4);
  float* bs = (float*)wsal(c, (size_t)C * 4);
  k_bias_only<<<(C + 255) / 256, 256, 0, c.s>>>(cb, sc, bs, C);
  return {sc, bs};
}

static float* conv_run(Ctx& c, const float* src, const float* w, SB sb,
                       const float* res, float* out,
                       int B, int Cin, int H, int W, int Cout,
                       int KH, int KW, int pad, int relu) {
  int Ktot = Cin * KH * KW;
  int Kpad = (Ktot + 31) & ~31;
  __bf16* wpk = (__bf16*)wsal(c, (size_t)Cout * Kpad * 2);
  int np = Cout * Kpad;
  k_pack_w<<<(np + 255) / 256, 256, 0, c.s>>>(w, wpk, Cout, Ktot, Kpad);
  if (!out) out = (float*)wsal(c, (size_t)B * Cout * H * W * 4);
  int M = B * H * W;
  dim3 g((M + 255) / 256, (Cout + 63) / 64);
  if (KH == 1 && KW == 1)
    conv_wmma<1><<<g, 256, 0, c.s>>>(src, wpk, sb.scale, sb.bias, res, out,
                                     B, Cin, H, W, Cout, KH, KW, pad, Kpad, relu);
  else if (KH == 3 && KW == 3)
    conv_wmma<9><<<g, 256, 0, c.s>>>(src, wpk, sb.scale, sb.bias, res, out,
                                     B, Cin, H, W, Cout, KH, KW, pad, Kpad, relu);
  else
    conv_wmma<0><<<g, 256, 0, c.s>>>(src, wpk, sb.scale, sb.bias, res, out,
                                     B, Cin, H, W, Cout, KH, KW, pad, Kpad, relu);
  return out;
}

static float* maxpool(Ctx& c, const float* in, int BC, int Ho, int Wo) {
  float* out = (float*)wsal(c, (size_t)BC * Ho * Wo * 4);
  int n = BC * Ho * Wo;
  k_maxpool2<<<(n + 255) / 256, 256, 0, c.s>>>(in, out, BC, Ho, Wo);
  return out;
}

static const float* bottleneck(Ctx& c, const float* xin, const Blk& p,
                               int B, int Cin, int H, int W, int planes) {
  int outc = planes * 2;
  SB s1 = fold(c, p.bn1, p.c1.b, planes);
  float* o1 = conv_run(c, xin, p.c1.w, s1, nullptr, nullptr, B, Cin, H, W, planes, 1, 1, 0, 1);
  SB s2 = fold(c, p.bn2, p.c2.b, planes);
  float* o2 = conv_run(c, o1, p.c2.w, s2, nullptr, nullptr, B, planes, H, W, planes, 3, 3, 1, 1);
  const float* idn = xin;
  if (p.has_ds) {
    SB sd = fold(c, p.dsbn, p.ds.b, outc);
    idn = conv_run(c, xin, p.ds.w, sd, nullptr, nullptr, B, Cin, H, W, outc, 1, 1, 0, 0);
  }
  SB s3 = fold(c, p.bn3, p.c3.b, outc);
  return conv_run(c, o2, p.c3.w, s3, idn, nullptr, B, planes, H, W, outc, 1, 1, 0, 1);
}

extern "C" void kernel_launch(void* const* d_in, const int* in_sizes, int n_in,
                              void* d_out, int out_size, void* d_ws, size_t ws_size,
                              hipStream_t stream) {
  (void)in_sizes; (void)n_in; (void)out_size; (void)ws_size;
  Ctx c{stream, (char*)d_ws, 0};

  // ---- walk inputs (jax pytree: dicts flattened in sorted-key order) ----
  int cur = 0;
  auto NX  = [&]() { return (const float*)d_in[cur++]; };
  auto rBN = [&](BNp& o) { o.b = NX(); o.g = NX(); o.m = NX(); o.v = NX(); };
  auto rCV = [&](CVp& o) { o.b = NX(); o.w = NX(); };
  auto rBK = [&](Blk& o, bool ds) {
    rBN(o.bn1); rBN(o.bn2); rBN(o.bn3);
    rCV(o.c1); rCV(o.c2); rCV(o.c3);
    o.has_ds = ds;
    if (ds) { rCV(o.ds); rBN(o.dsbn); }
  };

  const float* image = NX();
  BNp bn1; rBN(bn1);
  BNp bn2; rBN(bn2);
  const float* conv1_w = NX();
  CVp conv2; rCV(conv2);
  CVp convDa; rCV(convDa);
  CVp convDb; rCV(convDb);
  CVp convPa; rCV(convPa);
  CVp convPb; rCV(convPb);
  Blk L1[2]; rBK(L1[0], true);  rBK(L1[1], false);
  Blk L2[2]; rBK(L2[0], true);  rBK(L2[1], false);
  Blk L3[2]; rBK(L3[0], false); rBK(L3[1], false);

  const int B = 2, H0 = 768, W0 = 768;
  const int hc = 96, wc = 96;

  // ---- output regions (return order: dense_score, kpts_xy, kp_scores, desc)
  float* o_dense  = (float*)d_out;                       // B*768*768
  float* o_kpts   = o_dense + (size_t)B * H0 * W0;       // B*K*2
  float* o_scores = o_kpts + (size_t)B * KTOP * 2;       // B*K
  float* o_desc   = o_scores + (size_t)B * KTOP;         // B*K*128

  // ---- backbone ----
  SB sb1 = fold(c, bn1, nullptr, 64);
  float* x1 = conv_run(c, image, conv1_w, sb1, nullptr, nullptr, B, 3, H0, W0, 64, 7, 7, 3, 1);
  SB sb2 = fold(c, bn2, conv2.b, 64);
  float* x2 = conv_run(c, x1, conv2.w, sb2, nullptr, nullptr, B, 64, H0, W0, 64, 3, 3, 1, 1);
  const float* p1 = maxpool(c, x2, B * 64, 384, 384);

  const float* l1 = bottleneck(c, p1, L1[0], B, 64, 384, 384, 64);
  l1 = bottleneck(c, l1, L1[1], B, 128, 384, 384, 64);
  const float* p2 = maxpool(c, l1, B * 128, 192, 192);

  const float* l2 = bottleneck(c, p2, L2[0], B, 128, 192, 192, 128);
  l2 = bottleneck(c, l2, L2[1], B, 256, 192, 192, 128);
  const float* p3 = maxpool(c, l2, B * 256, 96, 96);

  const float* l3 = bottleneck(c, p3, L3[0], B, 256, 96, 96, 128);
  l3 = bottleneck(c, l3, L3[1], B, 256, 96, 96, 128);

  // ---- detector head ----
  SB spa = bias_only(c, convPa.b, 256);
  float* cPa = conv_run(c, l3, convPa.w, spa, nullptr, nullptr, B, 256, 96, 96, 256, 3, 3, 1, 1);
  SB spb = bias_only(c, convPb.b, 65);
  float* logits = conv_run(c, cPa, convPb.w, spb, nullptr, nullptr, B, 256, 96, 96, 65, 1, 1, 0, 0);
  {
    int n = B * hc * wc;
    k_softmax_shuffle<<<(n + 255) / 256, 256, 0, stream>>>(logits, o_dense, B, hc, wc);
  }

  // ---- descriptor head ----
  SB sda = bias_only(c, convDa.b, 256);
  float* cDa = conv_run(c, l3, convDa.w, sda, nullptr, nullptr, B, 256, 96, 96, 256, 3, 3, 1, 1);
  SB sdb = bias_only(c, convDb.b, 128);
  float* dd = conv_run(c, cDa, convDb.w, sdb, nullptr, nullptr, B, 256, 96, 96, 128, 1, 1, 0, 0);
  float* dmap = (float*)wsal(c, (size_t)B * 128 * 96 * 96 * 4);
  {
    int n = B * 96 * 96;
    k_l2norm<<<(n + 255) / 256, 256, 0, stream>>>(dd, dmap, B, 128, 96 * 96);
  }

  // ---- NMS (simple_nms, radius 8, 2 iterations) on o_dense ----
  const int NE = B * H0 * W0;
  const int GN = (NE + 255) / 256;
  float* scr  = (float*)wsal(c, (size_t)NE * 4);
  float* mpb  = (float*)wsal(c, (size_t)NE * 4);
  float* mask = (float*)wsal(c, (size_t)NE * 4);
  float* supp = (float*)wsal(c, (size_t)NE * 4);
  float* ss   = (float*)wsal(c, (size_t)NE * 4);
  float* maskd = (float*)wsal(c, (size_t)NE * 4);

  auto maxfilt = [&](const float* in, float* out, int r) {
    k_maxfilt_h<<<GN, 256, 0, stream>>>(in, scr, B, H0, W0, r);
    k_maxfilt_v<<<GN, 256, 0, stream>>>(scr, out, B, H0, W0, r);
  };

  maxfilt(o_dense, mpb, NMS_RAD);
  k_eq<<<GN, 256, 0, stream>>>(mask, o_dense, mpb, NE);
  for (int it = 0; it < 2; ++it) {
    maxfilt(mask, mpb, NMS_RAD);
    k_suppress<<<GN, 256, 0, stream>>>(mpb, o_dense, supp, ss, NE);
    maxfilt(ss, mpb, NMS_RAD);
    k_mask_update<<<GN, 256, 0, stream>>>(mask, ss, mpb, supp, NE);
  }
  k_border_masked<<<GN, 256, 0, stream>>>(maskd, mask, o_dense, B, H0, W0, BORDER_PX);

  // ---- subpixel refinement maps (7x7 separable) ----
  float* sumb = (float*)wsal(c, (size_t)NE * 4);
  float* dxb  = (float*)wsal(c, (size_t)NE * 4);
  float* dyb  = (float*)wsal(c, (size_t)NE * 4);
  k_box_h<<<GN, 256, 0, stream>>>(o_dense, scr, B, H0, W0, REF_RAD);
  k_box_v<<<GN, 256, 0, stream>>>(scr, sumb, B, H0, W0, REF_RAD);
  k_box_v<<<GN, 256, 0, stream>>>(o_dense, scr, B, H0, W0, REF_RAD);
  k_lin_h<<<GN, 256, 0, stream>>>(scr, dxb, B, H0, W0, REF_RAD);
  k_box_h<<<GN, 256, 0, stream>>>(o_dense, scr, B, H0, W0, REF_RAD);
  k_lin_v<<<GN, 256, 0, stream>>>(scr, dyb, B, H0, W0, REF_RAD);
  k_div_delta<<<GN, 256, 0, stream>>>(dyb, dxb, sumb, NE);

  // ---- top-K selection ----
  int* hist  = (int*)wsal(c, (size_t)B * 2048 * 4);
  int* tb    = (int*)wsal(c, (size_t)B * 4);
  int* above = (int*)wsal(c, (size_t)B * 4);
  int* cnt0  = (int*)wsal(c, (size_t)B * 4);
  int* cnt1  = (int*)wsal(c, (size_t)B * 4);
  int* kpidx = (int*)wsal(c, (size_t)B * KTOP * 4);
  hipMemsetAsync(hist, 0, (size_t)B * 2048 * 4, stream);
  hipMemsetAsync(cnt0, 0, (size_t)B * 4, stream);
  hipMemsetAsync(cnt1, 0, (size_t)B * 4, stream);
  k_hist<<<GN, 256, 0, stream>>>(maskd, hist, B, H0 * W0);
  k_select<<<1, 32, 0, stream>>>(hist, tb, above, B, KTOP);
  k_compact<<<GN, 256, 0, stream>>>(maskd, tb, above, cnt0, cnt1, o_scores, kpidx,
                                    B, H0 * W0, KTOP);

  // ---- keypoint refinement + descriptor sampling ----
  {
    int n = B * KTOP;
    k_refine<<<(n + 255) / 256, 256, 0, stream>>>(kpidx, dyb, dxb, o_kpts, B, KTOP, H0, W0);
    k_sample<<<n, 128, 0, stream>>>(o_kpts, dmap, o_desc, B, KTOP, 128, hc, wc, H0, W0);
  }
}
